// MultiHeadedAttention_26731876451176
// MI455X (gfx1250) — matrix operand order repaired and checked by reading, not compiled, not run on hardware
//
#include <hip/hip_runtime.h>

#define B_ 4
#define S_ 2048
#define D_ 1024
#define H_ 16
#define DEPTH_ 64
#define BS_ (B_ * S_)

typedef __attribute__((ext_vector_type(16))) __bf16 v16bf;
typedef __attribute__((ext_vector_type(8)))  __bf16 v8bf;
typedef __attribute__((ext_vector_type(8)))  float  v8f;
typedef __attribute__((ext_vector_type(4)))  float  v4f;

union FragAB { v16bf v; v8bf h[2]; };

// ---------------------------------------------------------------------------
// fp32 -> bf16 elementwise conversion
// ---------------------------------------------------------------------------
__global__ void cvt_f32_bf16(const float* __restrict__ src,
                             __bf16* __restrict__ dst, int n) {
    int i = blockIdx.x * blockDim.x + threadIdx.x;
    if (i < n) dst[i] = (__bf16)src[i];
}

// ---------------------------------------------------------------------------
// Register-blocked GEMM: each wave computes a 16x64 strip of
//   out[m,n] = sum_k A[m,k] * W[n,k] + bias[n]   (X @ W^T + b)
// A-fragment is loaded once per k-block and reused across 4 N-tiles
// (2.5 b128 loads per WMMA instead of 4).
// mode 0: store bf16, standard [BS, D] layout
// mode 1: store bf16, V-transposed layout [B, H, DEPTH, S]
// mode 2: store fp32, standard [BS, D] layout (final output)
// ---------------------------------------------------------------------------
__global__ void gemm16_bf16(const __bf16* __restrict__ A,
                            const __bf16* __restrict__ W,
                            const float* __restrict__ bias,
                            void* __restrict__ out, int mode) {
    const int lane = threadIdx.x & 31;
    const int half = lane >> 4;
    const int l15  = lane & 15;
    const int m0 = blockIdx.y * 16;
    const int n0 = blockIdx.x * 64;

    const __bf16* arow = A + (size_t)(m0 + l15) * D_;   // A frag: row = lane&15
    const __bf16* brow[4];
#pragma unroll
    for (int t = 0; t < 4; ++t)
        brow[t] = W + (size_t)(n0 + t * 16 + l15) * D_; // B frag: col = lane&15

    v8f acc0 = {}, acc1 = {}, acc2 = {}, acc3 = {};
    for (int kb = 0; kb < D_; kb += 32) {
        FragAB a;
        const int ka = kb + half * 16;         // A halves 0..15 = K ka..ka+15
        a.h[0] = *(const v8bf*)(arow + ka);
        a.h[1] = *(const v8bf*)(arow + ka + 8);
        const int kw = kb + half * 16;         // B halves 0..15 = K kw..kw+15
        FragAB b0, b1, b2, b3;
        b0.h[0] = *(const v8bf*)(brow[0] + kw);
        b0.h[1] = *(const v8bf*)(brow[0] + kw + 8);
        b1.h[0] = *(const v8bf*)(brow[1] + kw);
        b1.h[1] = *(const v8bf*)(brow[1] + kw + 8);
        b2.h[0] = *(const v8bf*)(brow[2] + kw);
        b2.h[1] = *(const v8bf*)(brow[2] + kw + 8);
        b3.h[0] = *(const v8bf*)(brow[3] + kw);
        b3.h[1] = *(const v8bf*)(brow[3] + kw + 8);
        acc0 = __builtin_amdgcn_wmma_f32_16x16x32_bf16(false, a.v, false, b0.v,
                                                       (short)0, acc0, false, false);
        acc1 = __builtin_amdgcn_wmma_f32_16x16x32_bf16(false, a.v, false, b1.v,
                                                       (short)0, acc1, false, false);
        acc2 = __builtin_amdgcn_wmma_f32_16x16x32_bf16(false, a.v, false, b2.v,
                                                       (short)0, acc2, false, false);
        acc3 = __builtin_amdgcn_wmma_f32_16x16x32_bf16(false, a.v, false, b3.v,
                                                       (short)0, acc3, false, false);
    }

    v8f accs[4] = {acc0, acc1, acc2, acc3};
#pragma unroll
    for (int t = 0; t < 4; ++t) {
        const int n = n0 + t * 16 + l15;
        const float bv = bias[n];
        if (mode == 0) {
            __bf16* o = (__bf16*)out;
#pragma unroll
            for (int j = 0; j < 8; ++j) {
                int row = m0 + j + 8 * half;
                o[(size_t)row * D_ + n] = (__bf16)(accs[t][j] + bv);
            }
        } else if (mode == 1) {
            __bf16* o = (__bf16*)out;
            const int h = n >> 6;                // head
            const int dd = n & (DEPTH_ - 1);     // depth index
#pragma unroll
            for (int j = 0; j < 8; ++j) {
                int row = m0 + j + 8 * half;     // global token row in [0, BS)
                int bb = row >> 11;              // / S_
                int ss = row & (S_ - 1);
                o[((size_t)(bb * H_ + h) * DEPTH_ + dd) * S_ + ss] =
                    (__bf16)(accs[t][j] + bv);
            }
        } else {
            float* o = (float*)out;
#pragma unroll
            for (int j = 0; j < 8; ++j) {
                int row = m0 + j + 8 * half;
                o[(size_t)row * D_ + n] = accs[t][j] + bv;
            }
        }
    }
}

// ---------------------------------------------------------------------------
// Fused attention for one (b, h, 16-query-row) block.
// 4 waves: scores (WMMA) -> LDS, softmax in LDS, attn weights -> global,
// then ctx = P @ V via WMMA reading P from LDS and V^T from global.
// ---------------------------------------------------------------------------
__global__ void __launch_bounds__(128)
attn16(const __bf16* __restrict__ Qbf, const __bf16* __restrict__ Kbf,
       const __bf16* __restrict__ Vt, const int* __restrict__ mask,
       float* __restrict__ attn, __bf16* __restrict__ concat) {
    __shared__ float sc[16][S_];      // 128 KB score strip
    __shared__ float red[16][8];
    __shared__ float rowmax[16];
    __shared__ float rowsum[16];

    const int tid  = threadIdx.x;
    const int lane = tid & 31;
    const int wave = tid >> 5;
    const int half = lane >> 4;
    const int l15  = lane & 15;
    const int m0 = blockIdx.x * 16;
    const int h  = blockIdx.y;
    const int b  = blockIdx.z;

    // ---- hoisted Q fragments (K dim = DEPTH = 64 -> two 32-wide k-blocks)
    const __bf16* qrow = Qbf + (size_t)(b * S_ + m0 + l15) * D_ + h * DEPTH_;
    FragAB aq0, aq1;
    {
        const int ka = half * 16;
        aq0.h[0] = *(const v8bf*)(qrow + ka);
        aq0.h[1] = *(const v8bf*)(qrow + ka + 8);
        aq1.h[0] = *(const v8bf*)(qrow + 32 + ka);
        aq1.h[1] = *(const v8bf*)(qrow + 32 + ka + 8);
    }

    // ---- scores: each wave handles 32 of the 128 key tiles
    for (int i = 0; i < 32; ++i) {
        const int n0 = (wave * 32 + i) * 16;
        const __bf16* krow =
            Kbf + (size_t)(b * S_ + n0 + l15) * D_ + h * DEPTH_ + half * 16;
        FragAB bk0, bk1;
        bk0.h[0] = *(const v8bf*)(krow);
        bk0.h[1] = *(const v8bf*)(krow + 8);
        bk1.h[0] = *(const v8bf*)(krow + 32);
        bk1.h[1] = *(const v8bf*)(krow + 40);
        v8f acc = {};
        acc = __builtin_amdgcn_wmma_f32_16x16x32_bf16(false, aq0.v, false, bk0.v,
                                                      (short)0, acc, false, false);
        acc = __builtin_amdgcn_wmma_f32_16x16x32_bf16(false, aq1.v, false, bk1.v,
                                                      (short)0, acc, false, false);
#pragma unroll
        for (int j = 0; j < 8; ++j) {
            const int m = j + 8 * half;
            const int col = n0 + l15;
            const int mv = mask[((size_t)b * S_ + m0 + m) * S_ + col];
            sc[m][col] = mv ? acc[j] * 0.125f : -1e9f;   // scale = 1/sqrt(64)
        }
    }
    __syncthreads();

    // ---- softmax over each of the 16 rows (8 threads per row)
    {
        const int row = tid & 15;
        const int slot = tid >> 4;
        const int c0 = slot * (S_ / 8);
        float mx = -3.0e38f;
        for (int c = c0; c < c0 + S_ / 8; ++c) mx = fmaxf(mx, sc[row][c]);
        red[row][slot] = mx;
        __syncthreads();
        if (slot == 0) {
            float m2 = red[row][0];
            for (int s2 = 1; s2 < 8; ++s2) m2 = fmaxf(m2, red[row][s2]);
            rowmax[row] = m2;
        }
        __syncthreads();
        const float rm = rowmax[row];
        float sum = 0.f;
        for (int c = c0; c < c0 + S_ / 8; ++c) {
            float e = __expf(sc[row][c] - rm);
            sc[row][c] = e;
            sum += e;
        }
        red[row][slot] = sum;
        __syncthreads();
        if (slot == 0) {
            float t = 0.f;
            for (int s2 = 0; s2 < 8; ++s2) t += red[row][s2];
            rowsum[row] = t;
        }
        __syncthreads();
    }

    // ---- normalize in LDS + coalesced store of attention weights
    for (int r = 0; r < 16; ++r) {
        const float inv = 1.0f / rowsum[r];
        float* dst = attn + ((size_t)(b * H_ + h) * S_ + m0 + r) * S_;
        for (int c = tid; c < S_; c += 128) {
            float p = sc[r][c] * inv;
            sc[r][c] = p;
            dst[c] = p;
        }
    }
    __syncthreads();

    // ---- ctx = P(16 x 2048) @ V(2048 x 64); one 16x16 n-tile per wave
    {
        const int n0 = wave * 16;
        const __bf16* vcol =
            Vt + ((size_t)(b * H_ + h) * DEPTH_ + n0 + l15) * S_ + half * 8;
        v8f acc = {};
        for (int kb = 0; kb < S_; kb += 32) {
            FragAB a, bb;
            const float* sp = &sc[l15][kb + half * 8];
            v4f f0 = *(const v4f*)(sp);
            v4f f1 = *(const v4f*)(sp + 4);
            v4f f2 = *(const v4f*)(sp + 16);
            v4f f3 = *(const v4f*)(sp + 20);
#pragma unroll
            for (int t = 0; t < 4; ++t) {
                a.v[t]      = (__bf16)f0[t];
                a.v[t + 4]  = (__bf16)f1[t];
                a.v[t + 8]  = (__bf16)f2[t];
                a.v[t + 12] = (__bf16)f3[t];
            }
            bb.h[0] = *(const v8bf*)(vcol + kb);
            bb.h[1] = *(const v8bf*)(vcol + kb + 16);
            acc = __builtin_amdgcn_wmma_f32_16x16x32_bf16(false, a.v, false, bb.v,
                                                          (short)0, acc, false, false);
        }
#pragma unroll
        for (int j = 0; j < 8; ++j) {
            const int srow = m0 + j + 8 * half;
            const int col = h * DEPTH_ + n0 + l15;
            concat[(size_t)(b * S_ + srow) * D_ + col] = (__bf16)acc[j];
        }
    }
}

// ---------------------------------------------------------------------------
extern "C" void kernel_launch(void* const* d_in, const int* in_sizes, int n_in,
                              void* d_out, int out_size, void* d_ws, size_t ws_size,
                              hipStream_t stream) {
    const float* query = (const float*)d_in[0];
    const float* key_  = (const float*)d_in[1];
    const float* value = (const float*)d_in[2];
    const int*   mask  = (const int*)d_in[3];
    const float* Wq = (const float*)d_in[4];
    const float* bq = (const float*)d_in[5];
    const float* Wk = (const float*)d_in[6];
    const float* bk = (const float*)d_in[7];
    const float* Wv = (const float*)d_in[8];
    const float* bv = (const float*)d_in[9];
    const float* Wo = (const float*)d_in[10];
    const float* bo = (const float*)d_in[11];

    char* ws = (char*)d_ws;
    size_t off = 0;
    auto alloc = [&](size_t bytes) -> char* {
        char* p = ws + off;
        off += (bytes + 255) & ~(size_t)255;
        return p;
    };
    const size_t actB = (size_t)BS_ * D_ * sizeof(__bf16);
    const size_t wB   = (size_t)D_ * D_ * sizeof(__bf16);
    __bf16* qin = (__bf16*)alloc(actB);
    __bf16* kin = (__bf16*)alloc(actB);
    __bf16* vin = (__bf16*)alloc(actB);
    __bf16* wqb = (__bf16*)alloc(wB);
    __bf16* wkb = (__bf16*)alloc(wB);
    __bf16* wvb = (__bf16*)alloc(wB);
    __bf16* wob = (__bf16*)alloc(wB);
    __bf16* Qb  = (__bf16*)alloc(actB);
    __bf16* Kb  = (__bf16*)alloc(actB);
    __bf16* Vt  = (__bf16*)alloc(actB);     // [B, H, DEPTH, S]
    __bf16* cc  = (__bf16*)alloc(actB);     // concat [BS, D]

    float* out  = (float*)d_out;
    float* attn = out + (size_t)BS_ * D_;   // tuple: (out, attn_weights)

    const int nAct = BS_ * D_;
    const int nW   = D_ * D_;
    cvt_f32_bf16<<<(nAct + 255) / 256, 256, 0, stream>>>(query, qin, nAct);
    cvt_f32_bf16<<<(nAct + 255) / 256, 256, 0, stream>>>(key_,  kin, nAct);
    cvt_f32_bf16<<<(nAct + 255) / 256, 256, 0, stream>>>(value, vin, nAct);
    cvt_f32_bf16<<<(nW + 255) / 256, 256, 0, stream>>>(Wq, wqb, nW);
    cvt_f32_bf16<<<(nW + 255) / 256, 256, 0, stream>>>(Wk, wkb, nW);
    cvt_f32_bf16<<<(nW + 255) / 256, 256, 0, stream>>>(Wv, wvb, nW);
    cvt_f32_bf16<<<(nW + 255) / 256, 256, 0, stream>>>(Wo, wob, nW);

    dim3 gGemm(D_ / 64, BS_ / 16);
    gemm16_bf16<<<gGemm, 32, 0, stream>>>(qin, wqb, bq, Qb, 0);
    gemm16_bf16<<<gGemm, 32, 0, stream>>>(kin, wkb, bk, Kb, 0);
    gemm16_bf16<<<gGemm, 32, 0, stream>>>(vin, wvb, bv, Vt, 1);

    dim3 gAttn(S_ / 16, H_, B_);
    attn16<<<gAttn, 128, 0, stream>>>(Qb, Kb, Vt, mask, attn, cc);

    gemm16_bf16<<<gGemm, 32, 0, stream>>>(cc, wob, bo, d_out, 2);
}
